// SceneGraphDenoiser_60138132078614
// MI455X (gfx1250) — matrix operand order repaired and checked by reading, not compiled, not run on hardware
//
#include <hip/hip_runtime.h>
#include <cstdint>
#include <cstddef>

#define BB   8
#define NN   96
#define DD   256
#define LL   4
#define NOBJN 151
#define NRELN 51
#define BN   (BB*NN)      // 768

typedef __attribute__((ext_vector_type(16))) __bf16 v16bf;
typedef __attribute__((ext_vector_type(8)))  __bf16 v8bf;
typedef __attribute__((ext_vector_type(8)))  float  v8f;

// ---------------- WMMA fragment helpers (contiguous k = 16*(lane/16) + elem, same in A and B) ----------
// A (16x32 bf16, row-major source with leading dim ld):
//   lanes 0-15 : M=lane,   elems 0..15 = K k0+0..15
//   lanes 16-31: M=lane-16, elems 0..15 = K k0+16..31
__device__ __forceinline__ v16bf frag_a_rm(const __bf16* tile, int ld, int k0) {
  const int lane = threadIdx.x & 31;
  const int row  = lane & 15;
  const int kg   = lane >> 4;
  const __bf16* p = tile + row * ld + k0 + (kg << 4);
  v8bf lo = *(const v8bf*)p;
  v8bf hi = *(const v8bf*)(p + 8);
  return __builtin_shufflevector(lo, hi, 0,1,2,3,4,5,6,7,8,9,10,11,12,13,14,15);
}

// B (32x16 bf16) from weights stored transposed (N x K row-major):
//   lane = output column n0+(lane&15); lanes 0-15 hold K k0..k0+15,
//   lanes 16-31 hold K k0+16..k0+31 (contiguous 32B per lane).
__device__ __forceinline__ v16bf frag_b_nk(const __bf16* wt, int ldk, int n0, int k0) {
  const int lane = threadIdx.x & 31;
  const int col  = lane & 15;
  const int kg   = lane >> 4;
  return *(const v16bf*)(wt + (size_t)(n0 + col) * ldk + k0 + (kg << 4));
}

__device__ __forceinline__ v8f wmma_bf16(v16bf a, v16bf b, v8f c) {
  return __builtin_amdgcn_wmma_f32_16x16x32_bf16(false, a, false, b, (short)0, c,
                                                 false, false);
}

__device__ __forceinline__ float silu_f(float x) {
  return x / (1.f + __expf(-x));
}

// ---------------- tiny scalar kernels ---------------------------------------

__global__ void k_temb(const int* __restrict__ t,
                       const float* __restrict__ tw1, const float* __restrict__ tb1,
                       const float* __restrict__ tw2, const float* __restrict__ tb2,
                       float* __restrict__ temb) {
  __shared__ float s0[BB][DD];
  __shared__ float s1[BB][DD];
  const int d = threadIdx.x;
  for (int b = 0; b < BB; ++b) {
    int half = d & 127;
    float freq = __expf(-logf(10000.f) * (float)half / 128.f);
    float arg = (float)t[b] * freq;
    s0[b][d] = (d < 128) ? __cosf(arg) : __sinf(arg);
  }
  __syncthreads();
  for (int b = 0; b < BB; ++b) {
    float acc = tb1[d];
    for (int k = 0; k < DD; ++k) acc += s0[b][k] * tw1[k * DD + d];
    s1[b][d] = silu_f(acc);
  }
  __syncthreads();
  for (int b = 0; b < BB; ++b) {
    float acc = tb2[d];
    for (int k = 0; k < DD; ++k) acc += s1[b][k] * tw2[k * DD + d];
    temb[b * DD + d] = acc;
  }
}

__global__ void k_init_h(const int* __restrict__ obj_t, const int* __restrict__ node_mask,
                         const float* __restrict__ obj_table, const float* __restrict__ temb,
                         float* __restrict__ h, __bf16* __restrict__ hbf) {
  const int idx = blockIdx.x * 256 + threadIdx.x;   // over BN*DD
  const int d = idx & (DD - 1);
  const int row = idx >> 8;
  const int b = row / NN;
  float v = (obj_table[(size_t)obj_t[row] * DD + d] + temb[b * DD + d]) *
            (float)node_mask[row];
  h[idx] = v;
  hbf[idx] = (__bf16)v;
}

// W (K x N f32, row-major) -> WT (N x K bf16, row-major)
__global__ void k_convT(const float* __restrict__ W, __bf16* __restrict__ WT,
                        int K, int N) {
  const int idx = blockIdx.x * 256 + threadIdx.x;
  if (idx >= K * N) return;
  const int k = idx % K, n = idx / K;
  WT[(size_t)n * K + k] = (__bf16)W[(size_t)k * N + n];
}

// zero-padded transpose for rel head W2 (256x51 -> 64x256 bf16)
__global__ void k_convT_pad(const float* __restrict__ W, __bf16* __restrict__ WT,
                            int K, int Nin, int Nout) {
  const int idx = blockIdx.x * 256 + threadIdx.x;
  if (idx >= K * Nout) return;
  const int k = idx % K, n = idx / K;
  WT[(size_t)n * K + k] = (n < Nin) ? (__bf16)W[(size_t)k * Nin + n] : (__bf16)0.f;
}

__global__ void k_conv_bf16(const float* __restrict__ X, __bf16* __restrict__ Y, int n) {
  const int idx = blockIdx.x * 256 + threadIdx.x;
  if (idx < n) Y[idx] = (__bf16)X[idx];
}

__global__ void k_silu_bias_bf16(const float* __restrict__ X, const float* __restrict__ bias,
                                 __bf16* __restrict__ Y, int n) {
  const int idx = blockIdx.x * 256 + threadIdx.x;
  if (idx >= n) return;
  float x = X[idx] + bias[idx & (DD - 1)];
  Y[idx] = (__bf16)silu_f(x);
}

__global__ void k_obj_head(const float* __restrict__ h, const float* __restrict__ W,
                           const float* __restrict__ bias, float* __restrict__ out) {
  const int idx = blockIdx.x * 256 + threadIdx.x;
  if (idx >= BN * NOBJN) return;
  const int o = idx % NOBJN, row = idx / NOBJN;
  float acc = bias[o];
  const float* hr = h + (size_t)row * DD;
  for (int k = 0; k < DD; ++k) acc += hr[k] * W[(size_t)k * NOBJN + o];
  out[idx] = acc;
}

__global__ void k_update_ln(float* __restrict__ h, const float* __restrict__ delta,
                            const float* __restrict__ ub2, const float* __restrict__ g,
                            const float* __restrict__ bb, const int* __restrict__ node_mask,
                            __bf16* __restrict__ hbf) {
  __shared__ float red[DD];
  const int row = blockIdx.x;
  const int d = threadIdx.x;
  const size_t o = (size_t)row * DD + d;
  float x = h[o] + delta[o] + ub2[d];
  red[d] = x;
  __syncthreads();
  for (int s = 128; s > 0; s >>= 1) { if (d < s) red[d] += red[d + s]; __syncthreads(); }
  float mu = red[0] * (1.f / DD);
  __syncthreads();
  float xc = x - mu;
  red[d] = xc * xc;
  __syncthreads();
  for (int s = 128; s > 0; s >>= 1) { if (d < s) red[d] += red[d + s]; __syncthreads(); }
  float var = red[0] * (1.f / DD);
  float y = (xc * rsqrtf(var + 1e-5f) * g[d] + bb[d]) * (float)node_mask[row];
  h[o] = y;
  hbf[o] = (__bf16)y;
}

// ---------------- generic WMMA GEMM: C[M,N] (+)= A[M,K]@W[K,N] ---------------
// A: bf16 row-major. WT: bf16 (N x K) transposed. M%16==0, K%32==0 (K<=256), N%16==0.
__global__ void __launch_bounds__(256) k_gemm(const __bf16* __restrict__ A,
                                              const __bf16* __restrict__ WT,
                                              float* __restrict__ C,
                                              int M, int K, int N, int accumulate) {
  __shared__ __align__(16) __bf16 sA[16 * DD];
  const int mt = blockIdx.x;
  const int tid = threadIdx.x;
  const int wave = tid >> 5;
  const int lane = tid & 31;
  const int col = lane & 15;
  const int rbase = (lane >> 4) << 3;
  {
    const __bf16* src = A + (size_t)mt * 16 * K;
    const int total = 16 * K;
    for (int idx = tid * 8; idx < total; idx += 256 * 8)
      *(v8bf*)(sA + idx) = *(const v8bf*)(src + idx);
  }
  __syncthreads();
  const int ntiles = N >> 4;
  for (int nt = wave; nt < ntiles; nt += 8) {
    v8f c = {};
    float* cbase = C + (size_t)mt * 16 * N + nt * 16 + col;
    if (accumulate) {
#pragma unroll
      for (int e = 0; e < 8; ++e) c[e] = cbase[(size_t)(rbase + e) * N];
    }
    for (int k0 = 0; k0 < K; k0 += 32) {
      v16bf a = frag_a_rm(sA, K, k0);
      v16bf b = frag_b_nk(WT, K, nt * 16, k0);
      c = wmma_bf16(a, b, c);
    }
#pragma unroll
    for (int e = 0; e < 8; ++e) cbase[(size_t)(rbase + e) * N] = c[e];
  }
}

// ------------- fused edge-message kernel: one block per (b,i) ----------------
// hid = silu(e_ji@W1b + hjpart[b,j] + b1); msgs = hid@W2 + b2;
// agg[b,i,:] = sum_j mask_t[b,i,j] * msgs[b,i,j,:]
__global__ void __launch_bounds__(256) k_edge_msg(
    const int* __restrict__ rel_t, const int* __restrict__ edge_mask,
    const __bf16* __restrict__ rel_tbl,           // NREL x D bf16
    const float* __restrict__ hjpart,             // BN x D
    const __bf16* __restrict__ W1T, const __bf16* __restrict__ W2T,  // D x D (N x K)
    const float* __restrict__ b1, const float* __restrict__ b2,
    float* __restrict__ agg) {
  __shared__ __align__(16) __bf16 sA[16 * DD];
  __shared__ __align__(16) __bf16 sH[16 * DD];
  const int bi = blockIdx.x;
  const int b = bi / NN, i = bi % NN;
  const int tid = threadIdx.x;
  const int wave = tid >> 5;
  const int lane = tid & 31;
  const int col = lane & 15;
  const int rbase = (lane >> 4) << 3;
  if (tid == 0) { __builtin_prefetch(W1T, 0, 1); __builtin_prefetch(W2T, 0, 1); }
  float aggAcc0 = 0.f, aggAcc1 = 0.f;
  for (int jt = 0; jt < NN / 16; ++jt) {
    __syncthreads();
    {  // stage e_ji tile: row r <- rel_table[rel_t[b, jt*16+r, i]]
      const int r = tid >> 4;
      const int cb = (tid & 15) << 4;
      const int j = jt * 16 + r;
      const int idx = rel_t[((b * NN) + j) * NN + i];
      *(v16bf*)(sA + r * DD + cb) = *(const v16bf*)(rel_tbl + (size_t)idx * DD + cb);
    }
    __syncthreads();
#pragma unroll
    for (int t = 0; t < 2; ++t) {   // GEMM1 -> hid (bf16 in sH)
      const int nt = wave + t * 8;
      v8f c = {};
      for (int k0 = 0; k0 < DD; k0 += 32) {
        v16bf a = frag_a_rm(sA, DD, k0);
        v16bf bf = frag_b_nk(W1T, DD, nt * 16, k0);
        c = wmma_bf16(a, bf, c);
      }
      const int n = nt * 16 + col;
      const float bias = b1[n];
#pragma unroll
      for (int e = 0; e < 8; ++e) {
        const int r = rbase + e;
        const int j = jt * 16 + r;
        float x = c[e] + hjpart[(size_t)(b * NN + j) * DD + n] + bias;
        sH[r * DD + n] = (__bf16)silu_f(x);
      }
    }
    __syncthreads();
#pragma unroll
    for (int t = 0; t < 2; ++t) {   // GEMM2 -> masked row-sum into agg regs
      const int nt = wave + t * 8;
      v8f c = {};
      for (int k0 = 0; k0 < DD; k0 += 32) {
        v16bf a = frag_a_rm(sH, DD, k0);
        v16bf bf = frag_b_nk(W2T, DD, nt * 16, k0);
        c = wmma_bf16(a, bf, c);
      }
      const int n = nt * 16 + col;
      const float bias = b2[n];
      float part = 0.f;
#pragma unroll
      for (int e = 0; e < 8; ++e) {
        const int j = jt * 16 + rbase + e;
        float m = (float)edge_mask[((b * NN) + j) * NN + i];  // mask_t[b,i,j]
        part += (c[e] + bias) * m;
      }
      if (t == 0) aggAcc0 += part; else aggAcc1 += part;
    }
  }
#pragma unroll
  for (int t = 0; t < 2; ++t) {
    float v = (t == 0) ? aggAcc0 : aggAcc1;
    v += __shfl_xor(v, 16, 32);
    if (lane < 16) agg[(size_t)bi * DD + (wave + t * 8) * 16 + col] = v;
  }
}

// ------------- fused relation head: one block per (b,i) ----------------------
__global__ void __launch_bounds__(256) k_rel_head(
    const int* __restrict__ rel_t, const __bf16* __restrict__ rel_tbl,
    const float* __restrict__ ra, const float* __restrict__ rb,   // BN x D
    const __bf16* __restrict__ WcT,   // D x D (N x K)
    const __bf16* __restrict__ W2T,   // 64 x D (padded N x K)
    const float* __restrict__ b1, const float* __restrict__ b2,
    float* __restrict__ out) {        // (B,N,N,NREL)
  __shared__ __align__(16) __bf16 sA[16 * DD];
  __shared__ __align__(16) __bf16 sH[16 * DD];
  const int bi = blockIdx.x;
  const int b = bi / NN, i = bi % NN;
  const int tid = threadIdx.x;
  const int wave = tid >> 5;
  const int lane = tid & 31;
  const int col = lane & 15;
  const int rbase = (lane >> 4) << 3;
  for (int jt = 0; jt < NN / 16; ++jt) {
    __syncthreads();
    {  // stage rel_e tile: row r <- rel_table[rel_t[b, i, jt*16+r]]
      const int r = tid >> 4;
      const int cb = (tid & 15) << 4;
      const int j = jt * 16 + r;
      const int idx = rel_t[((b * NN) + i) * NN + j];
      *(v16bf*)(sA + r * DD + cb) = *(const v16bf*)(rel_tbl + (size_t)idx * DD + cb);
    }
    __syncthreads();
#pragma unroll
    for (int t = 0; t < 2; ++t) {   // GEMM1: rel_e@Wc + ra_i + rb_j + b1 -> silu
      const int nt = wave + t * 8;
      v8f c = {};
      for (int k0 = 0; k0 < DD; k0 += 32) {
        v16bf a = frag_a_rm(sA, DD, k0);
        v16bf bf = frag_b_nk(WcT, DD, nt * 16, k0);
        c = wmma_bf16(a, bf, c);
      }
      const int n = nt * 16 + col;
      const float rai = ra[(size_t)bi * DD + n];
      const float bias = b1[n];
#pragma unroll
      for (int e = 0; e < 8; ++e) {
        const int r = rbase + e;
        const int j = jt * 16 + r;
        float x = c[e] + rai + rb[(size_t)(b * NN + j) * DD + n] + bias;
        sH[r * DD + n] = (__bf16)silu_f(x);
      }
    }
    __syncthreads();
    if (wave < 4) {                 // GEMM2: hid @ relh_w2 (padded to 64 cols)
      const int nt = wave;
      v8f c = {};
      for (int k0 = 0; k0 < DD; k0 += 32) {
        v16bf a = frag_a_rm(sH, DD, k0);
        v16bf bf = frag_b_nk(W2T, DD, nt * 16, k0);
        c = wmma_bf16(a, bf, c);
      }
      const int n = nt * 16 + col;
      if (n < NRELN) {
        const float bias = b2[n];
#pragma unroll
        for (int e = 0; e < 8; ++e) {
          const int j = jt * 16 + rbase + e;
          out[((size_t)(b * NN + i) * NN + j) * NRELN + n] = c[e] + bias;
        }
      }
    }
  }
}

// ---------------- host launcher ---------------------------------------------
extern "C" void kernel_launch(void* const* d_in, const int* in_sizes, int n_in,
                              void* d_out, int out_size, void* d_ws, size_t ws_size,
                              hipStream_t stream) {
  (void)in_sizes; (void)n_in; (void)out_size; (void)ws_size;
  const int*   obj_t     = (const int*)d_in[0];
  const int*   rel_t     = (const int*)d_in[1];
  const int*   t_in      = (const int*)d_in[2];
  const int*   node_mask = (const int*)d_in[3];
  const int*   edge_mask = (const int*)d_in[4];
  const float* obj_table = (const float*)d_in[5];
  const float* rel_table = (const float*)d_in[6];
  const float* tw1 = (const float*)d_in[7];
  const float* tb1 = (const float*)d_in[8];
  const float* tw2 = (const float*)d_in[9];
  const float* tb2 = (const float*)d_in[10];
  const float* msg_w1 = (const float*)d_in[11];
  const float* msg_b1 = (const float*)d_in[12];
  const float* msg_w2 = (const float*)d_in[13];
  const float* msg_b2 = (const float*)d_in[14];
  const float* upd_w1 = (const float*)d_in[15];
  const float* upd_b1 = (const float*)d_in[16];
  const float* upd_w2 = (const float*)d_in[17];
  const float* upd_b2 = (const float*)d_in[18];
  const float* ln_g = (const float*)d_in[19];
  const float* ln_b = (const float*)d_in[20];
  const float* objh_w = (const float*)d_in[21];
  const float* objh_b = (const float*)d_in[22];
  const float* relh_w1 = (const float*)d_in[23];
  const float* relh_b1 = (const float*)d_in[24];
  const float* relh_w2 = (const float*)d_in[25];
  const float* relh_b2 = (const float*)d_in[26];

  size_t off = 0;
  auto alloc = [&](size_t bytes) -> char* {
    char* p = (char*)d_ws + off;
    off += (bytes + 255) & ~(size_t)255;
    return p;
  };
  float*  temb  = (float*)alloc(BB * DD * 4);
  float*  h     = (float*)alloc((size_t)BN * DD * 4);
  __bf16* hbf   = (__bf16*)alloc((size_t)BN * DD * 2);
  float*  hjp   = (float*)alloc((size_t)BN * DD * 4);
  float*  agg   = (float*)alloc((size_t)BN * DD * 4);
  __bf16* aggbf = (__bf16*)alloc((size_t)BN * DD * 2);
  float*  t1    = (float*)alloc((size_t)BN * DD * 4);
  __bf16* t1h   = (__bf16*)alloc((size_t)BN * DD * 2);
  float*  delta = (float*)alloc((size_t)BN * DD * 4);
  float*  ra    = (float*)alloc((size_t)BN * DD * 4);
  float*  rbv   = (float*)alloc((size_t)BN * DD * 4);
  __bf16* reltb = (__bf16*)alloc((size_t)NRELN * DD * 2);
  const size_t WMAT = (size_t)DD * DD;        // 65536 elems
  __bf16* mw1aT = (__bf16*)alloc(LL * WMAT * 2);
  __bf16* mw1bT = (__bf16*)alloc(LL * WMAT * 2);
  __bf16* mw2T  = (__bf16*)alloc(LL * WMAT * 2);
  __bf16* uw1aT = (__bf16*)alloc(LL * WMAT * 2);
  __bf16* uw1bT = (__bf16*)alloc(LL * WMAT * 2);
  __bf16* uw2T  = (__bf16*)alloc(LL * WMAT * 2);
  __bf16* WaT   = (__bf16*)alloc(WMAT * 2);
  __bf16* WbT   = (__bf16*)alloc(WMAT * 2);
  __bf16* WcT   = (__bf16*)alloc(WMAT * 2);
  __bf16* W2rT  = (__bf16*)alloc((size_t)64 * DD * 2);

  float* obj_out = (float*)d_out;
  float* rel_out = obj_out + (size_t)BN * NOBJN;

  const int GCONV = (DD * DD + 255) / 256;  // 256 blocks

  k_temb<<<1, 256, 0, stream>>>(t_in, tw1, tb1, tw2, tb2, temb);
  k_init_h<<<BN, 256, 0, stream>>>(obj_t, node_mask, obj_table, temb, h, hbf);
  k_conv_bf16<<<(NRELN * DD + 255) / 256, 256, 0, stream>>>(rel_table, reltb, NRELN * DD);

  for (int l = 0; l < LL; ++l) {
    const float* mw1 = msg_w1 + (size_t)l * 2 * DD * DD;
    const float* mw2 = msg_w2 + (size_t)l * DD * DD;
    const float* uw1 = upd_w1 + (size_t)l * 2 * DD * DD;
    const float* uw2 = upd_w2 + (size_t)l * DD * DD;
    k_convT<<<GCONV, 256, 0, stream>>>(mw1,            mw1aT + l * WMAT, DD, DD);
    k_convT<<<GCONV, 256, 0, stream>>>(mw1 + DD * DD,  mw1bT + l * WMAT, DD, DD);
    k_convT<<<GCONV, 256, 0, stream>>>(mw2,            mw2T  + l * WMAT, DD, DD);
    k_convT<<<GCONV, 256, 0, stream>>>(uw1,            uw1aT + l * WMAT, DD, DD);
    k_convT<<<GCONV, 256, 0, stream>>>(uw1 + DD * DD,  uw1bT + l * WMAT, DD, DD);
    k_convT<<<GCONV, 256, 0, stream>>>(uw2,            uw2T  + l * WMAT, DD, DD);
  }
  k_convT<<<GCONV, 256, 0, stream>>>(relh_w1,               WaT, DD, DD);
  k_convT<<<GCONV, 256, 0, stream>>>(relh_w1 + DD * DD,     WbT, DD, DD);
  k_convT<<<GCONV, 256, 0, stream>>>(relh_w1 + 2 * DD * DD, WcT, DD, DD);
  k_convT_pad<<<(DD * 64 + 255) / 256, 256, 0, stream>>>(relh_w2, W2rT, DD, NRELN, 64);

  for (int l = 0; l < LL; ++l) {
    // hj_part = h @ mw1[:D]
    k_gemm<<<BN / 16, 256, 0, stream>>>(hbf, mw1aT + l * WMAT, hjp, BN, DD, DD, 0);
    // fused edge messages + aggregation
    k_edge_msg<<<BN, 256, 0, stream>>>(rel_t, edge_mask, reltb, hjp,
                                       mw1bT + l * WMAT, mw2T + l * WMAT,
                                       msg_b1 + l * DD, msg_b2 + l * DD, agg);
    // update MLP
    k_gemm<<<BN / 16, 256, 0, stream>>>(hbf, uw1aT + l * WMAT, t1, BN, DD, DD, 0);
    k_conv_bf16<<<BN, 256, 0, stream>>>(agg, aggbf, BN * DD);
    k_gemm<<<BN / 16, 256, 0, stream>>>(aggbf, uw1bT + l * WMAT, t1, BN, DD, DD, 1);
    k_silu_bias_bf16<<<BN, 256, 0, stream>>>(t1, upd_b1 + l * DD, t1h, BN * DD);
    k_gemm<<<BN / 16, 256, 0, stream>>>(t1h, uw2T + l * WMAT, delta, BN, DD, DD, 0);
    k_update_ln<<<BN, 256, 0, stream>>>(h, delta, upd_b2 + l * DD,
                                        ln_g + l * DD, ln_b + l * DD, node_mask, hbf);
  }

  // heads
  k_obj_head<<<(BN * NOBJN + 255) / 256, 256, 0, stream>>>(h, objh_w, objh_b, obj_out);
  k_gemm<<<BN / 16, 256, 0, stream>>>(hbf, WaT, ra, BN, DD, DD, 0);
  k_gemm<<<BN / 16, 256, 0, stream>>>(hbf, WbT, rbv, BN, DD, DD, 0);
  k_rel_head<<<BN, 256, 0, stream>>>(rel_t, reltb, ra, rbv, WcT, W2rT,
                                     relh_b1, relh_b2, rel_out);
}
